// GGNN_90031104459196
// MI455X (gfx1250) — compile-verified, matrix operand order checked by reading, not run on hardware
//
#include <hip/hip_runtime.h>
#include <hip/hip_bf16.h>

// MI455X / gfx1250 GGNN forward.
// Strategy: f16-input / f32-accumulate WMMA (v_wmma_f32_16x16x32_f16) for all
// dense GEMMs; weights pre-transformed once into per-lane WMMA B-fragment
// order; GRU gate GEMMs (2 x [N,64]x[64,192]) fused with the elementwise GRU
// update entirely in registers (no [N,192] scratch round-trip); edge
// scatter-add with float4 gathers + global_atomic_add_f32 (the HBM-bound part:
// ~1 GB traffic ~= 45us at 23.3 TB/s; GEMMs ~12 GFLOP are negligible on WMMA).

typedef __attribute__((ext_vector_type(16))) _Float16 v16h;
typedef __attribute__((ext_vector_type(8)))  float    v8f;

__device__ __forceinline__ v8f wmma16(v16h a, v16h b, v8f c) {
  // 8 args: (neg_a, A, neg_b, B, c_mod, C, reuse_a, reuse_b)
  return __builtin_amdgcn_wmma_f32_16x16x32_f16(false, a, false, b, (short)0, c,
                                                false, false);
}

__device__ __forceinline__ float sigmoidf_(float x) {
  return 1.0f / (1.0f + __expf(-x));
}

// ---------------------------------------------------------------------------
// Weight fragment prep: repack B matrices (f32) into f16 WMMA B-fragment order
// frag[((nt*2+ks)*32 + lane)*16 + j] = B[k(j,half,ks)][nt*16 + lane%16]
// with k(j,half,ks) = ks*32 + (j<8 ? half*8+j : 16 + half*8 + (j-8)).
// Regions: conv (2 layers x 4096), gi (12288), gh (12288), lin (4096).
// ---------------------------------------------------------------------------
__device__ __forceinline__ void frag_decode(int local, int& k, int& c) {
  int j    = local & 15;
  int lane = (local >> 4) & 31;
  int ks   = (local >> 9) & 1;
  int nt   = local >> 10;
  int half = lane >> 4;
  int jj   = (j < 8) ? (half * 8 + j) : (16 + half * 8 + (j - 8));
  k = ks * 32 + jj;
  c = nt * 16 + (lane & 15);
}

__global__ void prep_frags(const float* __restrict__ conv_w,
                           const float* __restrict__ w_ih,
                           const float* __restrict__ w_hh,
                           const float* __restrict__ lin_w,
                           _Float16* __restrict__ fragConv,
                           _Float16* __restrict__ fragGI,
                           _Float16* __restrict__ fragGH,
                           _Float16* __restrict__ fragLin) {
  int idx = blockIdx.x * blockDim.x + threadIdx.x;
  if (idx >= 36864) return;
  int k, c;
  if (idx < 8192) {                       // conv_w[layer][k][c], m = h @ W
    int layer = idx >> 12;
    frag_decode(idx & 4095, k, c);
    fragConv[idx] = (_Float16)conv_w[layer * 4096 + k * 64 + c];
  } else if (idx < 20480) {               // B[k][c] = w_ih[c][k] (gi = agg@W^T)
    int local = idx - 8192;
    frag_decode(local, k, c);
    fragGI[local] = (_Float16)w_ih[c * 64 + k];
  } else if (idx < 32768) {               // B[k][c] = w_hh[c][k]
    int local = idx - 20480;
    frag_decode(local, k, c);
    fragGH[local] = (_Float16)w_hh[c * 64 + k];
  } else {                                // B[k][c] = lin_w[c][k]
    int local = idx - 32768;
    frag_decode(local, k, c);
    fragLin[local] = (_Float16)lin_w[c * 64 + k];
  }
}

// ---------------------------------------------------------------------------
// Generic 64-col GEMM: C[N,64] = (relu? relu(A):A)[N,64] @ Bfrag (+ bias).
// One wave per 16-row stripe; 4 N-tiles x 2 K-steps = 8 WMMAs per wave.
// Requires nrows % 16 == 0 (N = 100000 -> 6250 stripes exactly).
// ---------------------------------------------------------------------------
__global__ void gemm64(const float* __restrict__ A,
                       const _Float16* __restrict__ Bfrag,
                       const float* __restrict__ bias,
                       float* __restrict__ C, int nrows, int relu) {
  int wave = (int)((blockIdx.x * blockDim.x + threadIdx.x) >> 5);
  int lane = threadIdx.x & 31;
  int row0 = wave * 16;
  if (row0 >= nrows) return;
  int half = lane >> 4, col = lane & 15, mrow = lane & 15;

  const float* rp = A + (size_t)(row0 + mrow) * 64 + half * 8;
  v16h a[2];
#pragma unroll
  for (int ks = 0; ks < 2; ++ks) {
    const float* p = rp + ks * 32;
#pragma unroll
    for (int j = 0; j < 8; ++j) {
      float v0 = p[j], v1 = p[16 + j];
      if (relu) { v0 = fmaxf(v0, 0.0f); v1 = fmaxf(v1, 0.0f); }
      a[ks][j]     = (_Float16)v0;
      a[ks][8 + j] = (_Float16)v1;
    }
  }
  const v16h* B = (const v16h*)Bfrag;
#pragma unroll
  for (int nt = 0; nt < 4; ++nt) {
    v8f acc = {};
#pragma unroll
    for (int ks = 0; ks < 2; ++ks)
      acc = wmma16(a[ks], B[(nt * 2 + ks) * 32 + lane], acc);
    float bv = bias ? bias[nt * 16 + col] : 0.0f;
#pragma unroll
    for (int v = 0; v < 8; ++v)
      C[(size_t)(row0 + half * 8 + v) * 64 + nt * 16 + col] = acc[v] + bv;
  }
}

// ---------------------------------------------------------------------------
// Fused GRU: gi = agg@W_ih^T+b_ih, gh = h@W_hh^T+b_hh (both 192-wide, in
// registers), gate math, h updated in place. One wave per 16-row stripe,
// 4 col-groups x 6 gates x 2 K-steps = 48 WMMAs per wave.
// r/z/n for channel c live in tiles tt, tt+4, tt+8 at identical lane/element.
// ---------------------------------------------------------------------------
__global__ void gru_fused(const float* __restrict__ agg,
                          float* __restrict__ h,
                          const _Float16* __restrict__ fragGI,
                          const _Float16* __restrict__ fragGH,
                          const float* __restrict__ b_ih,
                          const float* __restrict__ b_hh, int nrows) {
  int wave = (int)((blockIdx.x * blockDim.x + threadIdx.x) >> 5);
  int lane = threadIdx.x & 31;
  int row0 = wave * 16;
  if (row0 >= nrows) return;
  int half = lane >> 4, col = lane & 15, mrow = lane & 15;

  const float* rpa = agg + (size_t)(row0 + mrow) * 64 + half * 8;
  const float* rph = h   + (size_t)(row0 + mrow) * 64 + half * 8;
  v16h aA[2], aH[2];
#pragma unroll
  for (int ks = 0; ks < 2; ++ks) {
#pragma unroll
    for (int j = 0; j < 8; ++j) {
      aA[ks][j]     = (_Float16)rpa[ks * 32 + j];
      aA[ks][8 + j] = (_Float16)rpa[ks * 32 + 16 + j];
      aH[ks][j]     = (_Float16)rph[ks * 32 + j];
      aH[ks][8 + j] = (_Float16)rph[ks * 32 + 16 + j];
    }
  }
  const v16h* BI = (const v16h*)fragGI;
  const v16h* BH = (const v16h*)fragGH;
#pragma unroll
  for (int tt = 0; tt < 4; ++tt) {
    v8f ir = {}, iz = {}, in_ = {}, hr = {}, hz = {}, hn = {};
#pragma unroll
    for (int ks = 0; ks < 2; ++ks) {
      ir  = wmma16(aA[ks], BI[((tt    ) * 2 + ks) * 32 + lane], ir);
      iz  = wmma16(aA[ks], BI[((tt + 4) * 2 + ks) * 32 + lane], iz);
      in_ = wmma16(aA[ks], BI[((tt + 8) * 2 + ks) * 32 + lane], in_);
      hr  = wmma16(aH[ks], BH[((tt    ) * 2 + ks) * 32 + lane], hr);
      hz  = wmma16(aH[ks], BH[((tt + 4) * 2 + ks) * 32 + lane], hz);
      hn  = wmma16(aH[ks], BH[((tt + 8) * 2 + ks) * 32 + lane], hn);
    }
    int c = tt * 16 + col;
    float bir = b_ih[c], biz = b_ih[64 + c], bin = b_ih[128 + c];
    float bhr = b_hh[c], bhz = b_hh[64 + c], bhn = b_hh[128 + c];
#pragma unroll
    for (int v = 0; v < 8; ++v) {
      size_t off = (size_t)(row0 + half * 8 + v) * 64 + c;
      float hold = h[off];
      float rg = sigmoidf_(ir[v] + bir + hr[v] + bhr);
      float zg = sigmoidf_(iz[v] + biz + hz[v] + bhz);
      float ng = tanhf(in_[v] + bin + rg * (hn[v] + bhn));
      h[off] = (1.0f - zg) * ng + zg * hold;  // each wave owns its 16 rows
    }
  }
}

// ---------------------------------------------------------------------------
// Edge scatter-add: 16 threads/edge, float4 gather, 4 f32 atomics each.
// ---------------------------------------------------------------------------
__global__ void scatter_add(const float* __restrict__ m,
                            const int* __restrict__ src,
                            const int* __restrict__ dst,
                            float* __restrict__ agg, long long E) {
  long long tid = (long long)blockIdx.x * blockDim.x + threadIdx.x;
  long long e = tid >> 4;
  if (e >= E) return;
  int c = (int)(tid & 15) * 4;
  long long s = src[e], d = dst[e];
  float4 v = *(const float4*)(m + s * 64 + c);
  float* p = agg + d * 64 + c;
  atomicAdd(p + 0, v.x); atomicAdd(p + 1, v.y);
  atomicAdd(p + 2, v.z); atomicAdd(p + 3, v.w);
}

__global__ void fill_zero(float* __restrict__ p, long long n) {
  for (long long i = (long long)blockIdx.x * blockDim.x + threadIdx.x; i < n;
       i += (long long)gridDim.x * blockDim.x)
    p[i] = 0.0f;
}

// logits: s[n] = dot(y[n,:], attn_w) + attn_b ; one wave per row
__global__ void logits_k(const float* __restrict__ y,
                         const float* __restrict__ aw,
                         const float* __restrict__ ab,
                         float* __restrict__ s, int nrows) {
  int wave = (int)((blockIdx.x * blockDim.x + threadIdx.x) >> 5);
  int lane = threadIdx.x & 31;
  if (wave >= nrows) return;
  const float* yr = y + (size_t)wave * 64;
  float p = yr[lane] * aw[lane] + yr[lane + 32] * aw[lane + 32];
#pragma unroll
  for (int off = 16; off; off >>= 1) p += __shfl_down(p, off, 32);
  if (lane == 0) s[wave] = p + ab[0];
}

__global__ void reduce_max_partial(const float* __restrict__ s, int n,
                                   float* __restrict__ red) {
  __shared__ float sm[256];
  float m = -3.4e38f;
  for (int i = blockIdx.x * blockDim.x + threadIdx.x; i < n;
       i += gridDim.x * blockDim.x)
    m = fmaxf(m, s[i]);
  sm[threadIdx.x] = m; __syncthreads();
  for (int off = 128; off; off >>= 1) {
    if (threadIdx.x < off)
      sm[threadIdx.x] = fmaxf(sm[threadIdx.x], sm[threadIdx.x + off]);
    __syncthreads();
  }
  if (threadIdx.x == 0) red[blockIdx.x] = sm[0];
}

__global__ void reduce_max_final(const float* __restrict__ red,
                                 float* __restrict__ scal) {
  __shared__ float sm[256];
  sm[threadIdx.x] = red[threadIdx.x]; __syncthreads();
  for (int off = 128; off; off >>= 1) {
    if (threadIdx.x < off)
      sm[threadIdx.x] = fmaxf(sm[threadIdx.x], sm[threadIdx.x + off]);
    __syncthreads();
  }
  if (threadIdx.x == 0) scal[0] = sm[0];
}

__global__ void expsum_partial(const float* __restrict__ s, int n,
                               const float* __restrict__ scal,
                               float* __restrict__ red) {
  float M = scal[0];
  float acc = 0.0f;
  for (int i = blockIdx.x * blockDim.x + threadIdx.x; i < n;
       i += gridDim.x * blockDim.x)
    acc += __expf(s[i] - M);
  __shared__ float sm[256];
  sm[threadIdx.x] = acc; __syncthreads();
  for (int off = 128; off; off >>= 1) {
    if (threadIdx.x < off) sm[threadIdx.x] += sm[threadIdx.x + off];
    __syncthreads();
  }
  if (threadIdx.x == 0) red[blockIdx.x] = sm[0];
}

__global__ void reduce_sum_final(const float* __restrict__ red,
                                 float* __restrict__ scal,
                                 float* __restrict__ out) {
  __shared__ float sm[256];
  sm[threadIdx.x] = red[threadIdx.x]; __syncthreads();
  for (int off = 128; off; off >>= 1) {
    if (threadIdx.x < off) sm[threadIdx.x] += sm[threadIdx.x + off];
    __syncthreads();
  }
  if (threadIdx.x == 0) scal[1] = sm[0];
  if (threadIdx.x < 64) out[threadIdx.x] = 0.0f;  // zero output for atomics
}

// out[c] += sum_n softmax(s)[n] * y[n][c]; 4 row-groups x 64 channels / block
__global__ void weighted_sum(const float* __restrict__ y,
                             const float* __restrict__ s,
                             const float* __restrict__ scal,
                             float* __restrict__ out, int n) {
  int c = threadIdx.x & 63;
  int g = threadIdx.x >> 6;  // 0..3
  float M = scal[0], S = scal[1];
  float acc = 0.0f;
  for (int row = blockIdx.x * 4 + g; row < n; row += gridDim.x * 4)
    acc += __expf(s[row] - M) * y[(size_t)row * 64 + c];
  acc /= S;
  __shared__ float sm[256];
  sm[threadIdx.x] = acc; __syncthreads();
  if (g == 0)
    atomicAdd(&out[c], sm[c] + sm[64 + c] + sm[128 + c] + sm[192 + c]);
}

static inline size_t align_up(size_t x, size_t a) { return (x + a - 1) & ~(a - 1); }

extern "C" void kernel_launch(void* const* d_in, const int* in_sizes, int n_in,
                              void* d_out, int out_size, void* d_ws, size_t ws_size,
                              hipStream_t stream) {
  const float* x      = (const float*)d_in[0];
  const int*   edge   = (const int*)d_in[1];   // integer inputs -> const int*
  const float* conv_w = (const float*)d_in[2];
  const float* w_ih   = (const float*)d_in[3];
  const float* w_hh   = (const float*)d_in[4];
  const float* b_ih   = (const float*)d_in[5];
  const float* b_hh   = (const float*)d_in[6];
  const float* lin_w  = (const float*)d_in[7];
  const float* lin_b  = (const float*)d_in[8];
  const float* attn_w = (const float*)d_in[9];
  const float* attn_b = (const float*)d_in[10];
  float* out = (float*)d_out;

  const int n = in_sizes[0] / 64;              // nodes (100000, %16 == 0)
  const long long E = in_sizes[1] / 2;         // edges

  // workspace carve-up (all 64B aligned)
  size_t cur = 0;
  char* base = (char*)d_ws;
  float* h   = (float*)(base + cur); cur = align_up(cur + (size_t)n * 64 * 4, 64);
  float* m   = (float*)(base + cur); cur = align_up(cur + (size_t)n * 64 * 4, 64); // also y
  float* agg = (float*)(base + cur); cur = align_up(cur + (size_t)n * 64 * 4, 64);
  float* s   = (float*)(base + cur); cur = align_up(cur + (size_t)n * 4, 64);
  float* red = (float*)(base + cur); cur = align_up(cur + 512 * 4, 64);
  float* scal= (float*)(base + cur); cur = align_up(cur + 16 * 4, 64);
  _Float16* fragConv = (_Float16*)(base + cur); cur = align_up(cur + 8192  * 2, 64);
  _Float16* fragGI   = (_Float16*)(base + cur); cur = align_up(cur + 12288 * 2, 64);
  _Float16* fragGH   = (_Float16*)(base + cur); cur = align_up(cur + 12288 * 2, 64);
  _Float16* fragLin  = (_Float16*)(base + cur); cur = align_up(cur + 4096  * 2, 64);
  (void)ws_size; (void)n_in; (void)out_size;

  prep_frags<<<(36864 + 255) / 256, 256, 0, stream>>>(
      conv_w, w_ih, w_hh, lin_w, fragConv, fragGI, fragGH, fragLin);

  hipMemcpyAsync(h, x, (size_t)n * 64 * 4, hipMemcpyDeviceToDevice, stream);

  const int stripes = n / 16;                          // 6250 waves
  const int sblocks = (stripes * 32 + 255) / 256;      // 8 waves/block
  const long long sthreads = E * 16;
  const int scblocks = (int)((sthreads + 255) / 256);

  for (int layer = 0; layer < 2; ++layer) {
    gemm64<<<sblocks, 256, 0, stream>>>(h, fragConv + layer * 4096, nullptr, m, n, 0);
    fill_zero<<<4096, 256, 0, stream>>>(agg, (long long)n * 64);
    scatter_add<<<scblocks, 256, 0, stream>>>(m, edge, edge + E, agg, E);
    gru_fused<<<sblocks, 256, 0, stream>>>(agg, h, fragGI, fragGH, b_ih, b_hh, n);
  }

  // y = relu(h) @ lin_w^T + lin_b   (reuse m as y)
  gemm64<<<sblocks, 256, 0, stream>>>(h, fragLin, lin_b, m, n, 1);
  logits_k<<<(n * 32 + 255) / 256, 256, 0, stream>>>(m, attn_w, attn_b, s, n);
  reduce_max_partial<<<256, 256, 0, stream>>>(s, n, red);
  reduce_max_final<<<1, 256, 0, stream>>>(red, scal);
  expsum_partial<<<256, 256, 0, stream>>>(s, n, scal, red + 256);
  reduce_sum_final<<<1, 256, 0, stream>>>(red + 256, scal, out);
  weighted_sum<<<256, 256, 0, stream>>>(m, s, scal, out, n);
}
